// MeshCodec_3315714752542
// MI455X (gfx1250) — compile-verified, matrix operand order checked by reading, not run on hardware
//
#include <hip/hip_runtime.h>

// ============================================================================
// MeshCodec forward for gfx1250 (MI455X, wave32, WMMA).
//
// All dense GEMMs (sage Wp/Wl/Wr, conv1d-as-im2col-GEMM, final FC) run through
// one WMMA kernel using v_wmma_f32_16x16x32_bf16 (bf16 in, fp32 accum):
//  - 64x64 block tile, 8 wave32s, each wave owns a 16x32 C strip (2 WMMAs
//    per K-step, A fragment reused across both).
//  - Full-tile fast path stages tiles with float4 (global_load_b128) loads;
//    ragged edge tiles use a guarded scalar path.
//  - transA is a template parameter (no runtime address selects).
// Everything else (face geometry, embeddings, edge scatter-mean, l2norm,
// silu+layernorm, conditioning) is bandwidth-bound fp32 VALU work.
// ============================================================================

typedef __attribute__((ext_vector_type(16))) __bf16 bf16x16;
typedef __attribute__((ext_vector_type(8)))  float  f32x8;

__device__ __forceinline__ unsigned short f2bf(float f) {
  unsigned u = __float_as_uint(f);
  u += 0x7FFFu + ((u >> 16) & 1u);         // round-to-nearest-even
  return (unsigned short)(u >> 16);
}

// ---------------------------------------------------------------------------
// Generic GEMM: C[M,N] = op(A)[M,K] * W[K,N] (+bias) (+C if acc) (relu opt)
// A: fp32, lda given; TRANSA: A[m,k] = A[k*lda + m]. W row-major fp32 (K,N).
// ---------------------------------------------------------------------------
#define GBM 64
#define GBN 64
#define GBK 32
#define APAD 8   // row stride 40 halfs = 80B (keeps 16B-aligned fragment rows)
#define BPAD 8

template <int TRANSA>
__global__ __launch_bounds__(256)
void gemm_wmma_bf16(const float* __restrict__ A, const float* __restrict__ W,
                    const float* __restrict__ bias, float* __restrict__ C,
                    int M, int N, int K, int lda, int relu, int acc)
{
  __shared__ unsigned short sA[GBM][GBK + APAD];      // A tile, row-major (m,k)
  __shared__ unsigned short sBT[GBN][GBK + BPAD];     // B tile, transposed (n,k)

  const int tid   = threadIdx.x;
  const int lane  = tid & 31;
  const int wave  = tid >> 5;
  const int waveM = wave & 3;        // 0..3 -> 16-row strip
  const int waveN = wave >> 2;       // 0..1 -> 32-col strip
  const int row0  = blockIdx.x * GBM;
  const int col0  = blockIdx.y * GBN;

  f32x8 acc0 = {0.f, 0.f, 0.f, 0.f, 0.f, 0.f, 0.f, 0.f};
  f32x8 acc1 = {0.f, 0.f, 0.f, 0.f, 0.f, 0.f, 0.f, 0.f};

  const int half_sel = (lane >> 4) & 1;          // 0: lanes 0-15, 1: lanes 16-31
  const int mrow  = waveM * 16 + (lane & 15);    // A fragment row within tile
  const int ncol0 = waveN * 32 + (lane & 15);    // B fragment col (tile 0)

  const bool fullA = (!TRANSA) && (row0 + GBM <= M) && ((lda & 3) == 0);
  const bool fullB = (col0 + GBN <= N) && ((N & 3) == 0);

  for (int k0 = 0; k0 < K; k0 += GBK) {
    const bool fullK = (k0 + GBK <= K);

    // ---- stage A tile: 64x32 bf16 ----
    if (fullA && fullK) {
      #pragma unroll
      for (int i = 0; i < 2; ++i) {
        int r  = (tid >> 3) + i * 32;
        int c4 = (tid & 7) * 4;
        const float4 v = *(const float4*)(A + (size_t)(row0 + r) * lda + (k0 + c4));
        sA[r][c4 + 0] = f2bf(v.x);
        sA[r][c4 + 1] = f2bf(v.y);
        sA[r][c4 + 2] = f2bf(v.z);
        sA[r][c4 + 3] = f2bf(v.w);
      }
    } else {
      #pragma unroll
      for (int i = 0; i < 8; ++i) {
        int idx = tid + i * 256;          // 0..2047
        int r = idx >> 5;
        int k = idx & 31;
        int gr = row0 + r, gk = k0 + k;
        float v = 0.f;
        if (gr < M && gk < K)
          v = TRANSA ? A[(size_t)gk * lda + gr] : A[(size_t)gr * lda + gk];
        sA[r][k] = f2bf(v);
      }
    }
    // ---- stage B tile (transposed): 32x64 -> sBT[n][k] ----
    if (fullB && fullK) {
      #pragma unroll
      for (int i = 0; i < 2; ++i) {
        int k  = (tid >> 4) + i * 16;
        int n4 = (tid & 15) * 4;
        const float4 v = *(const float4*)(W + (size_t)(k0 + k) * N + (col0 + n4));
        sBT[n4 + 0][k] = f2bf(v.x);
        sBT[n4 + 1][k] = f2bf(v.y);
        sBT[n4 + 2][k] = f2bf(v.z);
        sBT[n4 + 3][k] = f2bf(v.w);
      }
    } else {
      #pragma unroll
      for (int i = 0; i < 8; ++i) {
        int idx = tid + i * 256;          // 0..2047
        int n = idx & 63;
        int k = idx >> 6;
        int gk = k0 + k, gn = col0 + n;
        float v = 0.f;
        if (gk < K && gn < N) v = W[(size_t)gk * N + gn];
        sBT[n][k] = f2bf(v);
      }
    }
    __syncthreads();

    // ---- build fragments per documented 16-bit VGPR layouts ----
    bf16x16 afrag, bfrag0, bfrag1;
    #pragma unroll
    for (int h = 0; h < 16; ++h) {
      int k = h + ((h >= 8) ? 8 : 0) + (half_sel ? 8 : 0);
      afrag[h]  = __builtin_bit_cast(__bf16, sA[mrow][k]);
      bfrag0[h] = __builtin_bit_cast(__bf16, sBT[ncol0][k]);
      bfrag1[h] = __builtin_bit_cast(__bf16, sBT[ncol0 + 16][k]);
    }
    acc0 = __builtin_amdgcn_wmma_f32_16x16x32_bf16(false, afrag, false, bfrag0,
                                                   (short)0, acc0, false, false);
    acc1 = __builtin_amdgcn_wmma_f32_16x16x32_bf16(false, afrag, false, bfrag1,
                                                   (short)0, acc1, false, false);
    __syncthreads();
  }

  // ---- store C: accum VGPR r holds (M = r + 8*half_sel, N = lane&15) ----
  const int crow0 = row0 + waveM * 16 + (half_sel ? 8 : 0);
  #pragma unroll
  for (int j = 0; j < 2; ++j) {
    const int ccol = col0 + waveN * 32 + j * 16 + (lane & 15);
    if (ccol < N) {
      const float bv = bias ? bias[ccol] : 0.f;
      #pragma unroll
      for (int r = 0; r < 8; ++r) {
        int crow = crow0 + r;
        if (crow < M) {
          float v = (j == 0 ? acc0[r] : acc1[r]) + bv;
          size_t off = (size_t)crow * N + ccol;
          if (acc) v += C[off];
          if (relu) v = fmaxf(v, 0.f);
          C[off] = v;
        }
      }
    }
  }
}

// ---------------------------------------------------------------------------
// Per-face geometry. raw layout (stride 24 floats):
// [0..8] fc (v0.xyz v1.xyz v2.xyz), [9..11] angles, [12] area,
// [13..15] normals, [16..18] inc (emangle), [19] emnoangle, [20] freq
// ---------------------------------------------------------------------------
#define RAW_STRIDE 24
#define PI_F 3.14159265358979323846f

__global__ void k_face_features(const float* __restrict__ verts,
                                const int* __restrict__ faces,
                                const float* __restrict__ theta,
                                const float* __restrict__ phi,
                                const float* __restrict__ freq,
                                float* __restrict__ raw,
                                int B, int Nv, int Nf)
{
  int n = blockIdx.x * blockDim.x + threadIdx.x;
  if (n >= B * Nf) return;
  int b = n / Nf;

  float fc[3][3];
  const int* fi = faces + (size_t)n * 3;
  #pragma unroll
  for (int i = 0; i < 3; ++i) {
    int vi = fi[i];
    vi = vi < 0 ? 0 : (vi >= Nv ? Nv - 1 : vi);
    const float* vp = verts + ((size_t)b * Nv + vi) * 3;
    fc[i][0] = vp[0]; fc[i][1] = vp[1]; fc[i][2] = vp[2];
  }

  float ev[3][3], nv[3][3];
  #pragma unroll
  for (int i = 0; i < 3; ++i) {
    int j = (i + 2) % 3;                          // roll(+1) along vertex axis
    float s = 0.f;
    #pragma unroll
    for (int c = 0; c < 3; ++c) { ev[i][c] = fc[i][c] - fc[j][c]; s += ev[i][c] * ev[i][c]; }
    float inv = 1.f / fmaxf(sqrtf(s), 1e-12f);
    #pragma unroll
    for (int c = 0; c < 3; ++c) nv[i][c] = ev[i][c] * inv;
  }

  float ang[3];
  #pragma unroll
  for (int c = 0; c < 3; ++c) {
    int c2 = (c + 2) % 3;                         // roll(+1) along coord axis
    float s = 0.f;
    #pragma unroll
    for (int i = 0; i < 3; ++i) s += nv[i][c] * nv[i][c2];
    float nd = -s;
    nd = fminf(fmaxf(nd, -1.f + 1e-5f), 1.f - 1e-5f);
    ang[c] = acosf(nd);
  }

  // cross(ev[0], ev[1])
  float cr0 = ev[0][1] * ev[1][2] - ev[0][2] * ev[1][1];
  float cr1 = ev[0][2] * ev[1][0] - ev[0][0] * ev[1][2];
  float cr2 = ev[0][0] * ev[1][1] - ev[0][1] * ev[1][0];
  float crn = sqrtf(cr0 * cr0 + cr1 * cr1 + cr2 * cr2);
  float area = crn * 0.5f;
  float inv = 1.f / fmaxf(crn, 1e-12f);
  float nx = cr0 * inv, ny = cr1 * inv, nz = cr2 * inv;

  float th = theta[b] * (PI_F / 180.f);
  float ph = phi[b]   * (PI_F / 180.f);
  float i0 = sinf(ph) * cosf(th), i1 = sinf(ph) * sinf(th), i2 = cosf(ph);
  float iin = 1.f / fmaxf(sqrtf(i0 * i0 + i1 * i1 + i2 * i2), 1e-12f);
  float nd2 = -(nx * i0 * iin + ny * i1 * iin + nz * i2 * iin);
  nd2 = fminf(fmaxf(nd2, -1.f + 1e-5f), 1.f - 1e-5f);
  float emno = acosf(nd2);

  float* r = raw + (size_t)n * RAW_STRIDE;
  #pragma unroll
  for (int i = 0; i < 3; ++i)
    #pragma unroll
    for (int c = 0; c < 3; ++c) r[i * 3 + c] = fc[i][c];
  r[9] = ang[0]; r[10] = ang[1]; r[11] = ang[2];
  r[12] = area;
  r[13] = nx; r[14] = ny; r[15] = nz;
  r[16] = i0; r[17] = i1; r[18] = i2;
  r[19] = emno;
  r[20] = freq[b];
}

// ---------------------------------------------------------------------------
// Embedding: feats[n, 0:1056] = concat of 7 tiny GEMVs (K <= 9).
// ---------------------------------------------------------------------------
__global__ void k_embed(const float* __restrict__ raw,
                        const float* cW, const float* cB,     // coor   9x576
                        const float* aW, const float* aB,     // angle  3x48
                        const float* arW, const float* arB,   // area   1x16
                        const float* nW, const float* nB,     // normal 3x192
                        const float* eaW, const float* eaB,   // emangle 3x192
                        const float* enW, const float* enB,   // emnoangle 1x16
                        const float* efW, const float* efB,   // emfreq 1x16
                        float* __restrict__ X, long long total)
{
  long long t = blockIdx.x * (long long)blockDim.x + threadIdx.x;
  if (t >= total) return;
  int n = (int)(t / 1056), d = (int)(t % 1056);
  const float* r = raw + (size_t)n * RAW_STRIDE;
  float v;
  if (d < 576) {
    v = cB[d];
    #pragma unroll
    for (int k = 0; k < 9; ++k) v += r[k] * cW[k * 576 + d];
  } else if (d < 624) {
    int dd = d - 576; v = aB[dd];
    #pragma unroll
    for (int k = 0; k < 3; ++k) v += r[9 + k] * aW[k * 48 + dd];
  } else if (d < 640) {
    int dd = d - 624; v = arB[dd] + r[12] * arW[dd];
  } else if (d < 832) {
    int dd = d - 640; v = nB[dd];
    #pragma unroll
    for (int k = 0; k < 3; ++k) v += r[13 + k] * nW[k * 192 + dd];
  } else if (d < 1024) {
    int dd = d - 832; v = eaB[dd];
    #pragma unroll
    for (int k = 0; k < 3; ++k) v += r[16 + k] * eaW[k * 192 + dd];
  } else if (d < 1040) {
    int dd = d - 1024; v = enB[dd] + r[19] * enW[dd];
  } else {
    int dd = d - 1040; v = efB[dd] + r[20] * efW[dd];
  }
  X[t] = v;
}

// ---------------------------------------------------------------------------
// Edge scatter: counts and segment sums (global fp32 atomics), then mean.
// ---------------------------------------------------------------------------
__global__ void k_seg_count(const int* __restrict__ dst, float* __restrict__ cnt, int E)
{
  int e = blockIdx.x * blockDim.x + threadIdx.x;
  if (e < E) atomicAdd(&cnt[dst[e]], 1.f);
}

__global__ void k_seg_sum(const float* __restrict__ H, const int* __restrict__ src,
                          const int* __restrict__ dst, float* __restrict__ S,
                          int E, int D, long long total)
{
  long long t = blockIdx.x * (long long)blockDim.x + threadIdx.x;
  if (t >= total) return;
  int e = (int)(t / D), d = (int)(t % D);
  atomicAdd(&S[(size_t)dst[e] * D + d], H[(size_t)src[e] * D + d]);
}

__global__ void k_seg_mean(float* __restrict__ S, const float* __restrict__ cnt,
                           int D, long long total)
{
  long long t = blockIdx.x * (long long)blockDim.x + threadIdx.x;
  if (t >= total) return;
  int n = (int)(t / D);
  S[t] /= fmaxf(cnt[n], 1.f);
}

// ---------------------------------------------------------------------------
// Row L2 normalize (block per row, LDS reduction).
// ---------------------------------------------------------------------------
__global__ __launch_bounds__(256)
void k_l2norm_rows(float* __restrict__ X, int D)
{
  __shared__ float red[256];
  float* row = X + (size_t)blockIdx.x * D;
  float ss = 0.f;
  for (int d = threadIdx.x; d < D; d += 256) { float v = row[d]; ss += v * v; }
  red[threadIdx.x] = ss;
  __syncthreads();
  for (int s = 128; s > 0; s >>= 1) {
    if (threadIdx.x < s) red[threadIdx.x] += red[threadIdx.x + s];
    __syncthreads();
  }
  float inv = 1.f / fmaxf(sqrtf(red[0]), 1e-12f);
  for (int d = threadIdx.x; d < D; d += 256) row[d] *= inv;
}

// ---------------------------------------------------------------------------
// y = LayerNorm(silu(x)) * g + b   (block per row; D <= 576)
// ---------------------------------------------------------------------------
__global__ __launch_bounds__(256)
void k_silu_layernorm(float* __restrict__ X, const float* __restrict__ g,
                      const float* __restrict__ b, int D)
{
  __shared__ float buf[576];
  __shared__ float rs[256], rq[256];
  float* row = X + (size_t)blockIdx.x * D;
  float s = 0.f, q = 0.f;
  for (int d = threadIdx.x; d < D; d += 256) {
    float x = row[d];
    float sl = x / (1.f + expf(-x));
    buf[d] = sl; s += sl; q += sl * sl;
  }
  rs[threadIdx.x] = s; rq[threadIdx.x] = q;
  __syncthreads();
  for (int st = 128; st > 0; st >>= 1) {
    if (threadIdx.x < st) { rs[threadIdx.x] += rs[threadIdx.x + st]; rq[threadIdx.x] += rq[threadIdx.x + st]; }
    __syncthreads();
  }
  float mean = rs[0] / D;
  float var  = rq[0] / D - mean * mean;
  float inv  = rsqrtf(var + 1e-5f);
  for (int d = threadIdx.x; d < D; d += 256)
    row[d] = (buf[d] - mean) * inv * g[d] + b[d];
}

// ---------------------------------------------------------------------------
// x[n,d] += freq[b]*Wf[d] + bf[d] + sum_j ang[b][j]*Wa[j,d] + ba[d]
// ---------------------------------------------------------------------------
__global__ void k_cond_add(float* __restrict__ X, const float* __restrict__ Wf,
                           const float* __restrict__ bf, const float* __restrict__ Wa,
                           const float* __restrict__ ba,
                           const float* __restrict__ theta, const float* __restrict__ phi,
                           const float* __restrict__ freq,
                           int Nf, int D, long long total)
{
  long long t = blockIdx.x * (long long)blockDim.x + threadIdx.x;
  if (t >= total) return;
  int n = (int)(t / D), d = (int)(t % D);
  int b = n / Nf;
  float th = theta[b] * (PI_F / 180.f), ph = phi[b] * (PI_F / 180.f);
  float add = freq[b] * Wf[d] + bf[d]
            + cosf(th) * Wa[0 * D + d] + sinf(th) * Wa[1 * D + d]
            + cosf(ph) * Wa[2 * D + d] + sinf(ph) * Wa[3 * D + d] + ba[d];
  X[t] += add;
}

// ---------------------------------------------------------------------------
// im2col for stride-10 conv: A2[(b*T+t), c*10+k] = X[(b*Nf + t*10 + k), c]
// ---------------------------------------------------------------------------
__global__ void k_im2col(const float* __restrict__ X, float* __restrict__ A2,
                         int Nf, int T, long long total)
{
  long long t = blockIdx.x * (long long)blockDim.x + threadIdx.x;
  if (t >= total) return;
  int m = (int)(t / 5760), j = (int)(t % 5760);
  int b = m / T, tt = m % T;
  int c = j / 10, k = j % 10;
  A2[t] = X[((size_t)b * Nf + (size_t)tt * 10 + k) * 576 + c];
}

// conv1d_W (256,576,10) -> Wc[(c*10+k), o]  (plain transpose; inner (576,10)
// flattens to exactly c*10+k)
__global__ void k_convw_t(const float* __restrict__ Ws, float* __restrict__ Wd, int total)
{
  int t = blockIdx.x * blockDim.x + threadIdx.x;
  if (t >= total) return;
  int j = t / 256, o = t % 256;
  Wd[t] = Ws[(size_t)o * 5760 + j];
}

// ===========================================================================
static inline unsigned cdiv(long long a, long long b) { return (unsigned)((a + b - 1) / b); }

extern "C" void kernel_launch(void* const* d_in, const int* in_sizes, int n_in,
                              void* d_out, int out_size, void* d_ws, size_t ws_size,
                              hipStream_t stream)
{
  (void)n_in; (void)out_size; (void)ws_size;

  // --- top-level inputs (setup_inputs dict insertion order) ---
  const float* vertices = (const float*)d_in[0];
  const int*   faces    = (const int*)d_in[1];
  const int*   edge     = (const int*)d_in[2];
  const float* theta    = (const float*)d_in[3];
  const float* phi      = (const float*)d_in[4];
  const float* freq     = (const float*)d_in[5];

  const int B    = in_sizes[3];
  const int Nv   = in_sizes[0] / (3 * B);
  const int Nf   = in_sizes[1] / (3 * B);
  const int E    = in_sizes[2] / 2;
  const int Ntot = B * Nf;
  const int* src = edge;
  const int* dst = edge + E;

  // --- params: assumed jax-pytree leaf order (nested dicts sorted by key) ---
  // params keys sorted: cond, conv1d_W, conv1d_b, emb, fc_W, fc_b, ln, sage
  // cond[i] keys sorted: Wa, Wf, ba, bf
  // emb keys sorted: angle_W, angle_b, area_W, area_b, coor_W, coor_b,
  //   emangle_W, emangle_b, emfreq_W, emfreq_b, emnoangle_W, emnoangle_b,
  //   normal_W, normal_b
  // sage[i] keys sorted: Wl, Wp, Wr, bl, bp
  const int P = 6;
  const float *condWa[4], *condWf[4], *condba[4], *condbf[4];
  for (int i = 0; i < 4; ++i) {
    condWa[i] = (const float*)d_in[P + 4 * i + 0];
    condWf[i] = (const float*)d_in[P + 4 * i + 1];
    condba[i] = (const float*)d_in[P + 4 * i + 2];
    condbf[i] = (const float*)d_in[P + 4 * i + 3];
  }
  const float* conv1d_W = (const float*)d_in[22];
  const float* conv1d_b = (const float*)d_in[23];
  const float* angle_W  = (const float*)d_in[24];
  const float* angle_b  = (const float*)d_in[25];
  const float* area_W   = (const float*)d_in[26];
  const float* area_b   = (const float*)d_in[27];
  const float* coor_W   = (const float*)d_in[28];
  const float* coor_b   = (const float*)d_in[29];
  const float* emangle_W   = (const float*)d_in[30];
  const float* emangle_b   = (const float*)d_in[31];
  const float* emfreq_W    = (const float*)d_in[32];
  const float* emfreq_b    = (const float*)d_in[33];
  const float* emnoangle_W = (const float*)d_in[34];
  const float* emnoangle_b = (const float*)d_in[35];
  const float* normal_W = (const float*)d_in[36];
  const float* normal_b = (const float*)d_in[37];
  const float* fc_W = (const float*)d_in[38];
  const float* fc_b = (const float*)d_in[39];
  const float *ln_g[5], *ln_b[5];
  for (int i = 0; i < 5; ++i) {
    ln_g[i] = (const float*)d_in[40 + 2 * i];
    ln_b[i] = (const float*)d_in[41 + 2 * i];
  }
  const float *sWl[5], *sWp[5], *sWr[5], *sbl[5], *sbp[5];
  for (int i = 0; i < 5; ++i) {
    sWl[i] = (const float*)d_in[50 + 5 * i + 0];
    sWp[i] = (const float*)d_in[50 + 5 * i + 1];
    sWr[i] = (const float*)d_in[50 + 5 * i + 2];
    sbl[i] = (const float*)d_in[50 + 5 * i + 3];
    sbp[i] = (const float*)d_in[50 + 5 * i + 4];
  }

  const int NFC = in_sizes[39];                 // 4050
  const int KFC = in_sizes[38] / NFC;           // 2250
  const int T   = (Nf - 10) / 10 + 1;           // 2250 conv output length
  const int MC  = B * T;                        // 4500 im2col rows

  // --- workspace layout ---
  size_t off = 0;
  auto wsalloc = [&](size_t bytes) {
    void* p = (char*)d_ws + off;
    off += (bytes + 255) & ~(size_t)255;
    return p;
  };
  float* RAW = (float*)wsalloc((size_t)Ntot * RAW_STRIDE * 4);
  float* X0  = (float*)wsalloc((size_t)Ntot * 1056 * 4);
  float* X1  = (float*)wsalloc((size_t)Ntot * 1056 * 4);
  float* H   = (float*)wsalloc((size_t)Ntot * 1056 * 4);   // also reused for im2col
  float* S   = (float*)wsalloc((size_t)Ntot * 1056 * 4);
  float* CNT = (float*)wsalloc((size_t)Ntot * 4);
  float* WC  = (float*)wsalloc((size_t)5760 * 256 * 4);
  float* Y   = (float*)wsalloc((size_t)MC * 256 * 4);

  auto gemm = [&](const float* A, const float* W, const float* bias, float* C,
                  int M, int N, int K, int lda, int relu, int acc) {
    dim3 grid(cdiv(M, GBM), cdiv(N, GBN));
    gemm_wmma_bf16<0><<<grid, dim3(256), 0, stream>>>(A, W, bias, C, M, N, K, lda,
                                                      relu, acc);
  };
  auto gemmT = [&](const float* A, const float* W, const float* bias, float* C,
                   int M, int N, int K, int lda) {
    dim3 grid(cdiv(M, GBM), cdiv(N, GBN));
    gemm_wmma_bf16<1><<<grid, dim3(256), 0, stream>>>(A, W, bias, C, M, N, K, lda,
                                                      0, 0);
  };

  // 1) face geometry + raw features
  k_face_features<<<cdiv(Ntot, 256), 256, 0, stream>>>(vertices, faces, theta, phi,
                                                       freq, RAW, B, Nv, Nf);
  // 2) embedding -> X0 (Ntot x 1056)
  {
    long long tot = (long long)Ntot * 1056;
    k_embed<<<cdiv(tot, 256), 256, 0, stream>>>(RAW, coor_W, coor_b, angle_W, angle_b,
        area_W, area_b, normal_W, normal_b, emangle_W, emangle_b,
        emnoangle_W, emnoangle_b, emfreq_W, emfreq_b, X0, tot);
  }
  // 3) edge in-degree counts (reused every layer)
  hipMemsetAsync(CNT, 0, (size_t)Ntot * 4, stream);
  k_seg_count<<<cdiv(E, 256), 256, 0, stream>>>(dst, CNT, E);

  // 4) five SAGE layers with cond-add before layers 1..4
  const int dims[6] = {1056, 64, 128, 256, 256, 576};
  float* cur = X0;
  float* nxt = X1;
  for (int i = 0; i < 5; ++i) {
    const int Din = dims[i], Dout = dims[i + 1];
    const long long totIn = (long long)Ntot * Din;
    if (i > 0)
      k_cond_add<<<cdiv(totIn, 256), 256, 0, stream>>>(cur, condWf[i - 1], condbf[i - 1],
          condWa[i - 1], condba[i - 1], theta, phi, freq, Nf, Din, totIn);

    // h = relu(x @ Wp + bp)
    gemm(cur, sWp[i], sbp[i], H, Ntot, Din, Din, Din, 1, 0);
    // mean aggregation over edges
    hipMemsetAsync(S, 0, (size_t)Ntot * Din * 4, stream);
    k_seg_sum<<<cdiv((long long)E * Din, 256), 256, 0, stream>>>(H, src, dst, S, E, Din,
                                                                 (long long)E * Din);
    k_seg_mean<<<cdiv(totIn, 256), 256, 0, stream>>>(S, CNT, Din, totIn);
    // out = mean @ Wl + bl + x @ Wr ; l2norm ; silu+ln
    gemm(S,   sWl[i], sbl[i],  nxt, Ntot, Dout, Din, Din, 0, 0);
    gemm(cur, sWr[i], nullptr, nxt, Ntot, Dout, Din, Din, 0, 1);
    k_l2norm_rows<<<Ntot, 256, 0, stream>>>(nxt, Dout);
    k_silu_layernorm<<<Ntot, 256, 0, stream>>>(nxt, ln_g[i], ln_b[i], Dout);
    float* tmp = cur; cur = nxt; nxt = tmp;
  }

  // 5) conv1d as im2col GEMM: Y[(b,t), o]
  {
    long long tot = (long long)MC * 5760;
    k_im2col<<<cdiv(tot, 256), 256, 0, stream>>>(cur, H, Nf, T, tot);
    k_convw_t<<<cdiv(5760 * 256, 256), 256, 0, stream>>>(conv1d_W, WC, 5760 * 256);
    gemm(H, WC, conv1d_b, Y, MC, 256, 5760, 5760, 0, 0);
  }

  // 6) final FC contracts the t axis: out[b,o,:] = Y[b,:,o]^T @ fc_W + fc_b
  float* OUT = (float*)d_out;
  for (int b = 0; b < B; ++b)
    gemmT(Y + (size_t)b * T * 256, fc_W, fc_b, OUT + (size_t)b * 256 * NFC,
          256, NFC, KFC, 256);
}
